// NextWordLSTM_62002147885324
// MI455X (gfx1250) — compile-verified, matrix operand order checked by reading, not run on hardware
//
#include <hip/hip_runtime.h>
#include <hip/hip_bf16.h>
#include <math.h>

typedef __attribute__((ext_vector_type(16))) _Float16 v16h;
typedef __attribute__((ext_vector_type(8)))  float    v8f;

#define V_SZ 30522
#define E_SZ 128
#define H_SZ 256
#define B_SZ 1024
#define T_SZ 256
#define G4H  1024   /* 4*H */
#define KCAT 384    /* H + E : [h | emb] concatenated K */
#define NBLK 64     /* persistent recurrence blocks (B/16) */

// ---------------------------------------------------------------------------
// Fragment loader for wave32 WMMA 16-bit A/B layout (ISA 7.12.2):
//   lanes 0-15 : row/col = lane,    K-pairs {0..7, 16..23}
//   lanes 16-31: row/col = lane-16, K-pairs {8..15,24..31}
// VGPR q holds K = ((q&4)?16:0) + kb + 2*(q&3) .. +1   (kb = 0 or 8)
// Two contiguous 16B runs per lane -> compiler emits paired global_load_b128.
// ---------------------------------------------------------------------------
__device__ __forceinline__ v16h load_frag16(const _Float16* __restrict__ row, int kb) {
  v16h f;
#pragma unroll
  for (int q = 0; q < 8; ++q) {
    int kl = ((q & 4) << 2) + kb + 2 * (q & 3);
    f[2 * q]     = row[kl];
    f[2 * q + 1] = row[kl + 1];
  }
  return f;
}

// ------------------------------- prep kernels ------------------------------
__global__ void prep_emb_kernel(const float* __restrict__ emb,
                                _Float16* __restrict__ emb16, int n) {
  int i = blockIdx.x * blockDim.x + threadIdx.x;
  if (i < n) emb16[i] = (i < E_SZ) ? (_Float16)0.f : (_Float16)emb[i]; // padding_idx row 0
}

__global__ void prep_wcat_kernel(const float* __restrict__ wih,
                                 const float* __restrict__ whh,
                                 _Float16* __restrict__ wcat) {
  int i = blockIdx.x * blockDim.x + threadIdx.x;
  if (i < G4H * KCAT) {
    int g = i / KCAT, k = i % KCAT;
    float v = (k < H_SZ) ? whh[g * H_SZ + k] : wih[g * E_SZ + (k - H_SZ)];
    wcat[i] = (_Float16)v;
  }
}

__global__ void prep_bias_kernel(const float* __restrict__ bih,
                                 const float* __restrict__ bhh,
                                 float* __restrict__ bias) {
  int i = blockIdx.x * blockDim.x + threadIdx.x;
  if (i < G4H) bias[i] = bih[i] + bhh[i];
}

__global__ void prep_fcw_kernel(const float* __restrict__ fcw,
                                _Float16* __restrict__ fcw16, int n) {
  int i = blockIdx.x * blockDim.x + threadIdx.x;
  if (i < n) fcw16[i] = (_Float16)fcw[i];
}

__global__ void init_state_kernel(_Float16* __restrict__ h16,
                                  float* __restrict__ c32,
                                  _Float16* __restrict__ hl16,
                                  unsigned* __restrict__ barrier_ctr) {
  int i = blockIdx.x * blockDim.x + threadIdx.x;
  if (i == 0) *barrier_ctr = 0u;                 // re-armed every call -> replay-safe
  if (i < B_SZ * H_SZ) {
    h16[i] = (_Float16)0.f;
    c32[i] = 0.f;
    hl16[i] = (_Float16)0.f;
  }
}

// ---------------------------------------------------------------------------
// Persistent LSTM recurrence: one launch, internal loop over t with a
// monotonic grid barrier. Each block owns 16 batch rows exclusively; the
// barrier only orders h16 (the block's own A-fragment K<256 reads) across steps.
// gates = [h | emb[x[:,t]]] @ [W_hh | W_ih]^T  (WMMA f16, K=384)
// ---------------------------------------------------------------------------
__global__ __launch_bounds__(256) void lstm_persistent_kernel(
    const _Float16* __restrict__ emb16, const _Float16* __restrict__ wcat,
    const float* __restrict__ bias, const int* __restrict__ x,
    const int* __restrict__ lengths,
    _Float16* __restrict__ h16, float* __restrict__ c32,
    _Float16* __restrict__ hlast16, unsigned* __restrict__ barrier_ctr) {
  __shared__ float gates[16][G4H];                 // 64 KB: 16 b-rows x 4H
  const int b0   = blockIdx.x * 16;
  const int tid  = threadIdx.x;
  const int lane = tid & 31;
  const int wave = tid >> 5;
  const int m    = lane & 15;
  const int kb   = (lane >> 4) * 8;
  const int brow = b0 + m;
  const int gsel = lane & 15;
  const int j    = tid;                            // pointwise hidden column
  const float bi = bias[j], bf = bias[H_SZ + j];
  const float bg = bias[2 * H_SZ + j], bo = bias[3 * H_SZ + j];
  const int my_len_m1 = lengths[b0 + (tid & 15)];  // not used; keep loads simple

#pragma unroll 1
  for (int t = 0; t < T_SZ; ++t) {
    // ---- A fragments: [ h(b,0:256) | emb16[x[b,t]](0:128) ], 12 K-chunks ----
    const int token = x[brow * T_SZ + t];
    v16h afrag[12];
#pragma unroll
    for (int kc = 0; kc < 8; ++kc)
      afrag[kc] = load_frag16(h16 + brow * H_SZ + kc * 32, kb);
#pragma unroll
    for (int kc = 0; kc < 4; ++kc)
      afrag[8 + kc] = load_frag16(emb16 + (size_t)token * E_SZ + kc * 32, kb);

    // ---- each wave: N-columns [wave*128, wave*128+128), 8 tiles of 16 ----
#pragma unroll 1
    for (int nt = 0; nt < 8; ++nt) {
      const int n0 = wave * 128 + nt * 16;
      v16h bfrag[12];                              // preload -> load/WMMA overlap
#pragma unroll
      for (int kc = 0; kc < 12; ++kc)
        bfrag[kc] = load_frag16(wcat + (size_t)(n0 + gsel) * KCAT + kc * 32, kb);
      v8f acc = {};
#pragma unroll
      for (int kc = 0; kc < 12; ++kc)
        acc = __builtin_amdgcn_wmma_f32_16x16x32_f16(false, afrag[kc], false, bfrag[kc],
                                                     (short)0, acc, false, false);
#pragma unroll
      for (int r = 0; r < 8; ++r)
        gates[r + ((lane >> 4) << 3)][n0 + gsel] = acc[r];   // D layout -> LDS
    }
    __syncthreads();

    // ---- pointwise cell update: thread tid owns column j for all 16 rows ----
#pragma unroll 1
    for (int r = 0; r < 16; ++r) {
      const int b = b0 + r;
      float ig = gates[r][j] + bi;
      float fg = gates[r][H_SZ + j] + bf;
      float gg = gates[r][2 * H_SZ + j] + bg;
      float og = gates[r][3 * H_SZ + j] + bo;
      float si = 1.f / (1.f + __expf(-ig));
      float sf = 1.f / (1.f + __expf(-fg));
      float so = 1.f / (1.f + __expf(-og));
      float c  = sf * c32[b * H_SZ + j] + si * tanhf(gg);
      float h  = so * tanhf(c);
      c32[b * H_SZ + j] = c;
      h16[b * H_SZ + j] = (_Float16)h;
      if (t == lengths[b] - 1) hlast16[b * H_SZ + j] = (_Float16)h;
    }
    __syncthreads();                               // LDS reuse + all stores issued

    // ---- grid barrier (monotonic counter; release h16 stores, then acquire) ----
    if (t != T_SZ - 1) {
      if (tid == 0) {
        __threadfence();                           // push h16 stores to L2 (global_wb)
        __hip_atomic_fetch_add(barrier_ctr, 1u, __ATOMIC_RELEASE,
                               __HIP_MEMORY_SCOPE_AGENT);
        const unsigned target = (unsigned)NBLK * (unsigned)(t + 1);
        while (__hip_atomic_load(barrier_ctr, __ATOMIC_ACQUIRE,
                                 __HIP_MEMORY_SCOPE_AGENT) < target)
          __builtin_amdgcn_s_sleep(1);
      }
      __syncthreads();
      __threadfence();                             // invalidate stale WGP$ lines
    }
  }
  (void)my_len_m1;
}

// ---------------------------------------------------------------------------
// out = h_last @ fc_W^T + fc_b   (M=1024, N=30522, K=256) — WMMA f16.
// Each wave owns one N-tile; B fragments register-resident across 64 M-tiles.
// ---------------------------------------------------------------------------
__global__ __launch_bounds__(256) void fc_kernel(
    const _Float16* __restrict__ hlast16, const _Float16* __restrict__ fcw16,
    const float* __restrict__ fcb, float* __restrict__ out) {
  const int tid  = threadIdx.x;
  const int lane = tid & 31;
  const int wave = tid >> 5;
  const int tile = blockIdx.x * 8 + wave;
  const int NT   = (V_SZ + 15) / 16;               // 1908 tiles
  if (tile >= NT) return;                          // wave-uniform guard
  const int n0   = tile * 16;
  const int kb   = (lane >> 4) * 8;
  const int nsel = lane & 15;
  int gclamp = n0 + nsel; if (gclamp >= V_SZ) gclamp = V_SZ - 1;

  v16h bfrag[8];
#pragma unroll
  for (int kc = 0; kc < 8; ++kc)
    bfrag[kc] = load_frag16(fcw16 + (size_t)gclamp * H_SZ + kc * 32, kb);

  const int n = n0 + nsel;
  const float bias_n = (n < V_SZ) ? fcb[n] : 0.f;

#pragma unroll 1
  for (int mt = 0; mt < B_SZ / 16; ++mt) {
    const int m0 = mt * 16;
    v16h af[8];
#pragma unroll
    for (int kc = 0; kc < 8; ++kc)
      af[kc] = load_frag16(hlast16 + (size_t)(m0 + (lane & 15)) * H_SZ + kc * 32, kb);
    v8f acc = {};
#pragma unroll
    for (int kc = 0; kc < 8; ++kc)
      acc = __builtin_amdgcn_wmma_f32_16x16x32_f16(false, af[kc], false, bfrag[kc],
                                                   (short)0, acc, false, false);
    if (n < V_SZ) {
#pragma unroll
      for (int r = 0; r < 8; ++r) {
        const int b = m0 + r + ((lane >> 4) << 3);
        out[(size_t)b * V_SZ + n] = acc[r] + bias_n;
      }
    }
  }
}

// ---------------------------------------------------------------------------
// Bug-faithful repetition penalty: id = argmax_e emb[x[b,t]] (first max),
// out[b,id] += count * -log(RP). One wave per batch row; deterministic.
// ---------------------------------------------------------------------------
__global__ __launch_bounds__(32) void penalty_kernel(
    const int* __restrict__ x, const int* __restrict__ lengths,
    const float* __restrict__ emb, float* __restrict__ out) {
  __shared__ int counts[E_SZ];
  const int b = blockIdx.x;
  const int lane = threadIdx.x;
#pragma unroll
  for (int j = 0; j < 4; ++j) counts[lane + 32 * j] = 0;
  __syncthreads();

  const int len = lengths[b];
  for (int t = 0; t < len; ++t) {
    const int token = x[b * T_SZ + t];               // wave-uniform
    if (token == 0) continue;                        // zero row -> argmax 0 -> excluded
    const float* row = emb + (size_t)token * E_SZ;
    float bv = row[lane];
    int   bi = lane;
#pragma unroll
    for (int j = 1; j < 4; ++j) {
      float v = row[lane + 32 * j];
      if (v > bv) { bv = v; bi = lane + 32 * j; }    // ascending idx: '>' keeps first max
    }
#pragma unroll
    for (int off = 16; off >= 1; off >>= 1) {
      float ov = __shfl_xor(bv, off, 32);
      int   oi = __shfl_xor(bi, off, 32);
      if (ov > bv || (ov == bv && oi < bi)) { bv = ov; bi = oi; }
    }
    if (lane == 0 && bi != 0) counts[bi]++;
  }
  __syncthreads();

  const float pen = -logf(1.1f);
#pragma unroll
  for (int j = 0; j < 4; ++j) {
    const int id = lane + 32 * j;
    const int c  = counts[id];
    if (c > 0 && id != 0) out[(size_t)b * V_SZ + id] += (float)c * pen;
  }
}

// ---------------------------------------------------------------------------
extern "C" void kernel_launch(void* const* d_in, const int* in_sizes, int n_in,
                              void* d_out, int out_size, void* d_ws, size_t ws_size,
                              hipStream_t stream) {
  const int*   x    = (const int*)d_in[0];
  const int*   lens = (const int*)d_in[1];
  const float* emb  = (const float*)d_in[2];
  const float* wih  = (const float*)d_in[3];
  const float* whh  = (const float*)d_in[4];
  const float* bih  = (const float*)d_in[5];
  const float* bhh  = (const float*)d_in[6];
  const float* fcw  = (const float*)d_in[7];
  const float* fcb  = (const float*)d_in[8];
  float* out = (float*)d_out;

  // workspace layout (~26 MB)
  char* ws = (char*)d_ws;
  size_t off = 0;
  _Float16* emb16 = (_Float16*)(ws + off); off += (size_t)V_SZ * E_SZ * sizeof(_Float16);
  _Float16* wcat  = (_Float16*)(ws + off); off += (size_t)G4H * KCAT * sizeof(_Float16);
  _Float16* fcw16 = (_Float16*)(ws + off); off += (size_t)V_SZ * H_SZ * sizeof(_Float16);
  float*    bias  = (float*)   (ws + off); off += (size_t)G4H * sizeof(float);
  _Float16* h16   = (_Float16*)(ws + off); off += (size_t)B_SZ * H_SZ * sizeof(_Float16);
  float*    c32   = (float*)   (ws + off); off += (size_t)B_SZ * H_SZ * sizeof(float);
  _Float16* hl16  = (_Float16*)(ws + off); off += (size_t)B_SZ * H_SZ * sizeof(_Float16);
  unsigned* bctr  = (unsigned*)(ws + off); off += 256;
  (void)ws_size; (void)in_sizes; (void)n_in; (void)out_size;

  // prepack f16 operands + init state/barrier (idempotent -> replay-deterministic)
  prep_emb_kernel <<<(V_SZ * E_SZ + 255) / 256, 256, 0, stream>>>(emb, emb16, V_SZ * E_SZ);
  prep_wcat_kernel<<<(G4H * KCAT + 255) / 256, 256, 0, stream>>>(wih, whh, wcat);
  prep_bias_kernel<<<(G4H + 255) / 256,        256, 0, stream>>>(bih, bhh, bias);
  prep_fcw_kernel <<<(V_SZ * H_SZ + 255) / 256, 256, 0, stream>>>(fcw, fcw16, V_SZ * H_SZ);
  init_state_kernel<<<(B_SZ * H_SZ + 255) / 256, 256, 0, stream>>>(h16, c32, hl16, bctr);

  // whole recurrence: single persistent launch (64 blocks, internal grid barrier)
  lstm_persistent_kernel<<<NBLK, 256, 0, stream>>>(emb16, wcat, bias, x, lens,
                                                   h16, c32, hl16, bctr);

  // classifier + penalty
  const int NT = (V_SZ + 15) / 16;                 // 1908
  fc_kernel<<<(NT + 7) / 8, 256, 0, stream>>>(hl16, fcw16, fcb, out);
  penalty_kernel<<<B_SZ, 32, 0, stream>>>(x, lens, emb, out);
}
